// PatternMemory_81071802679341
// MI455X (gfx1250) — compile-verified, matrix operand order checked by reading, not run on hardware
//
#include <hip/hip_runtime.h>
#include <hip/hip_bf16.h>

typedef __attribute__((ext_vector_type(16))) _Float16 v16h;
typedef __attribute__((ext_vector_type(8)))  float    v8f;
typedef __attribute__((ext_vector_type(4)))  float    f32x4;

#define B_DIM 8
#define T_DIM 1024
#define C_DIM 1024
#define P_DIM 100
#define CTX_PAD 16    // ctx_h padded rows (1 tile)
#define TRG_PAD 112   // trg_h padded rows (7 tiles)
#define LAMBDA 0.1f
#define SIM_THRESHOLD 0.5f
#define QPT 4   // float4 quads per thread in apply kernel (ILP / MLP)

// ---------------------------------------------------------------------------
// Kernel 1: H = X @ Wslice via v_wmma_f32_16x16x32_f16 (f32 accumulate).
//   grid.y == 0 : X = context  (8 rows, pad to 16),  W = W1[0:C]    -> ctx_h
//   grid.y >= 1 : X = triggers (100 rows, pad 112),  W = W1[C:2C]   -> trg_h
// A-frag (16x32 f16, ISA 7.12.2): lane l -> row M = l&15, khalf = (l>>4)*8,
//   elems 0..7 -> K = khalf+e, elems 8..15 -> K = khalf+16+(e-8).
// Row padding: clamp row pointer (unconditional b128 loads) and multiply by
// a 0/1 lane mask. Output is stored to PADDED H (invalid rows are exactly 0
// because their A rows were masked), so stores are unconditional too.
// ---------------------------------------------------------------------------
__global__ __launch_bounds__(32)
void pm_gemm_h_kernel(const float* __restrict__ context,
                      const float* __restrict__ triggers,
                      const float* __restrict__ W1,
                      float* __restrict__ ctx_h,
                      float* __restrict__ trg_h) {
  const int lane   = threadIdx.x;
  const int l15    = lane & 15;
  const int half16 = lane >> 4;
  const int khalf  = half16 * 8;
  const int nBase  = blockIdx.x * 16;

  const float* X; const float* W; float* H; int mBase, mValid;
  if (blockIdx.y == 0) {
    X = context;  W = W1;                           H = ctx_h;
    mBase = 0;                                      mValid = B_DIM;
  } else {
    X = triggers; W = W1 + (size_t)C_DIM * C_DIM;   H = trg_h;
    mBase = ((int)blockIdx.y - 1) * 16;             mValid = P_DIM;
  }

  const int   mRow    = mBase + l15;
  const bool  mOK     = (mRow < mValid);
  const int   safeRow = mOK ? mRow : 0;
  const float msk     = mOK ? 1.0f : 0.0f;
  const f32x4* __restrict__ xrow4 = (const f32x4*)(X + (size_t)safeRow * C_DIM);
  const float* __restrict__ wcol  = W + nBase + l15;

  v8f acc = {};
  for (int k0 = 0; k0 < C_DIM; k0 += 32) {
    const int kq = (k0 + khalf) >> 2;               // f32x4 index of first chunk
    f32x4 x0 = xrow4[kq]     * msk;                 // K = khalf+0..3
    f32x4 x1 = xrow4[kq + 1] * msk;                 // K = khalf+4..7
    f32x4 x2 = xrow4[kq + 4] * msk;                 // K = khalf+16..19
    f32x4 x3 = xrow4[kq + 5] * msk;                 // K = khalf+20..23

    v16h a, b;
    a[0]=(_Float16)x0.x; a[1]=(_Float16)x0.y; a[2]=(_Float16)x0.z; a[3]=(_Float16)x0.w;
    a[4]=(_Float16)x1.x; a[5]=(_Float16)x1.y; a[6]=(_Float16)x1.z; a[7]=(_Float16)x1.w;
    a[8]=(_Float16)x2.x; a[9]=(_Float16)x2.y; a[10]=(_Float16)x2.z; a[11]=(_Float16)x2.w;
    a[12]=(_Float16)x3.x; a[13]=(_Float16)x3.y; a[14]=(_Float16)x3.z; a[15]=(_Float16)x3.w;

#pragma unroll
    for (int e = 0; e < 16; ++e) {
      const int k = k0 + khalf + e + ((e >= 8) ? 8 : 0);
      b[e] = (_Float16)wcol[(size_t)k * C_DIM];     // strided column loads of W
    }
    // 8 args: (neg_a, A, neg_b, B, c_mod, C, reuse_a, reuse_b)
    acc = __builtin_amdgcn_wmma_f32_16x16x32_f16(false, a, false, b,
                                                 (short)0, acc, false, false);
  }

  // Unconditional stores into padded H (invalid rows hold zeros).
  float* __restrict__ hbase = H + (size_t)(mBase + half16 * 8) * C_DIM + nBase + l15;
#pragma unroll
  for (int r = 0; r < 8; ++r) {
    hbase[(size_t)r * C_DIM] = acc[r];
  }
}

// ---------------------------------------------------------------------------
// Kernel 2: per pattern p -> logits over 8 batches, sigmoid-mean, threshold.
// ---------------------------------------------------------------------------
__global__ __launch_bounds__(256)
void pm_score_kernel(const float* __restrict__ ctx_h,
                     const float* __restrict__ trg_h,
                     const float* __restrict__ b1,
                     const float* __restrict__ W2,
                     const float* __restrict__ b2,
                     const float* __restrict__ conf,
                     float* __restrict__ weights) {
  const int p   = blockIdx.x;
  const int tid = threadIdx.x;
  __shared__ float red[256];
  __shared__ float ssum_sh;
  if (tid == 0) ssum_sh = 0.0f;

  float part[B_DIM];
#pragma unroll
  for (int b = 0; b < B_DIM; ++b) part[b] = 0.0f;

  for (int c = tid; c < C_DIM; c += 256) {
    const float t  = trg_h[(size_t)p * C_DIM + c] + b1[c];
    const float w2 = W2[c];
#pragma unroll
    for (int b = 0; b < B_DIM; ++b) {
      const float h = ctx_h[(size_t)b * C_DIM + c] + t;
      part[b] += (h > 0.0f ? h : 0.0f) * w2;
    }
  }

  for (int b = 0; b < B_DIM; ++b) {
    __syncthreads();
    red[tid] = part[b];
    __syncthreads();
    for (int s = 128; s > 0; s >>= 1) {
      if (tid < s) red[tid] += red[tid + s];
      __syncthreads();
    }
    if (tid == 0) {
      const float logit = red[0] + b2[0];
      ssum_sh += 1.0f / (1.0f + __expf(-logit));
    }
  }
  __syncthreads();
  if (tid == 0) {
    const float score = ssum_sh / (float)B_DIM;
    weights[p] = (score > SIM_THRESHOLD) ? score * conf[p] : 0.0f;
  }
}

// ---------------------------------------------------------------------------
// Kernel 2b: compact active patterns (weight != 0) -> branch-free apply loop.
// Serial scan by one lane: 100 iters, negligible; preserves pattern order.
// ---------------------------------------------------------------------------
__global__ void pm_compact_kernel(const float* __restrict__ weights,
                                  float* __restrict__ actW,
                                  int* __restrict__ actIdx,
                                  int* __restrict__ nact) {
  if (threadIdx.x == 0 && blockIdx.x == 0) {
    int n = 0;
    for (int p = 0; p < P_DIM; ++p) {
      const float wp = weights[p];
      if (wp != 0.0f) { actW[n] = wp; actIdx[n] = p; ++n; }
    }
    nact[0] = n;
    for (int p = n; p < P_DIM; ++p) { actW[p] = 0.0f; actIdx[p] = 0; }  // deterministic
  }
}

// ---------------------------------------------------------------------------
// Kernel 3 (bandwidth-dominant, ~490 MB):
//   acc[q] = sum_active wp * biases[p, quad]   (NT b128 streaming loads)
//   out[b] = attn[b] + 0.1*acc                 (amortized over 8 batches)
// QPT=4 quads/thread + p-unroll 2 => up to 8 independent loads in flight.
// ---------------------------------------------------------------------------
__global__ __launch_bounds__(256)
void pm_apply_kernel(const float* __restrict__ attn,
                     const float* __restrict__ biases,
                     const float* __restrict__ actW,
                     const int* __restrict__ actIdx,
                     const int* __restrict__ nactp,
                     float* __restrict__ out) {
  const int tid = threadIdx.x;
  __shared__ float sW[P_DIM];
  __shared__ int   sI[P_DIM];
  if (tid < P_DIM) { sW[tid] = actW[tid]; sI[tid] = actIdx[tid]; }
  __syncthreads();

  const int    nact  = nactp[0];
  const size_t nQuad = (size_t)T_DIM * T_DIM / 4;
  const size_t base  = (size_t)blockIdx.x * (256 * QPT) + tid;  // grid sized exactly

  const f32x4* __restrict__ b4 = (const f32x4*)biases;
  f32x4 acc[QPT];
#pragma unroll
  for (int q = 0; q < QPT; ++q) acc[q] = (f32x4){0.f, 0.f, 0.f, 0.f};

#pragma unroll 2
  for (int pi = 0; pi < nact; ++pi) {
    const float  wp = sW[pi];
    const size_t pb = (size_t)sI[pi] * nQuad;
#pragma unroll
    for (int q = 0; q < QPT; ++q) {
      const f32x4 v = __builtin_nontemporal_load(&b4[pb + base + (size_t)q * 256]);
      acc[q] += wp * v;
    }
  }

  const f32x4* __restrict__ a4 = (const f32x4*)attn;
  f32x4* __restrict__ o4 = (f32x4*)out;
#pragma unroll
  for (int b = 0; b < B_DIM; ++b) {
#pragma unroll
    for (int q = 0; q < QPT; ++q) {
      const size_t i = (size_t)b * nQuad + base + (size_t)q * 256;
      const f32x4 av = __builtin_nontemporal_load(&a4[i]);
      __builtin_nontemporal_store(av + LAMBDA * acc[q], &o4[i]);
    }
  }
}

// ---------------------------------------------------------------------------
extern "C" void kernel_launch(void* const* d_in, const int* in_sizes, int n_in,
                              void* d_out, int out_size, void* d_ws, size_t ws_size,
                              hipStream_t stream) {
  const float* attn     = (const float*)d_in[0];  // (8,1024,1024)
  const float* context  = (const float*)d_in[1];  // (8,1024)
  const float* triggers = (const float*)d_in[2];  // (100,1024)
  const float* biases   = (const float*)d_in[3];  // (100,1024,1024)
  const float* conf     = (const float*)d_in[4];  // (100,)
  const float* W1       = (const float*)d_in[5];  // (2048,1024)
  const float* b1       = (const float*)d_in[6];  // (1024,)
  const float* W2       = (const float*)d_in[7];  // (1024,)
  const float* b2       = (const float*)d_in[8];  // (1,)
  float* out = (float*)d_out;

  float* ws      = (float*)d_ws;
  float* ctx_h   = ws;                                   // CTX_PAD*1024 (rows 8..15 zero)
  float* trg_h   = ws + (size_t)CTX_PAD * C_DIM;         // TRG_PAD*1024 (rows 100..111 zero)
  float* weights = trg_h + (size_t)TRG_PAD * C_DIM;      // 100
  float* actW    = weights + P_DIM;                      // 100
  int*   actIdx  = (int*)(actW + P_DIM);                 // 100
  int*   nact    = actIdx + P_DIM;                       // 1

  // Kernel 1: both GEMMs. grid = (N tiles = 64, M tiles = 1 ctx + 7 trig)
  dim3 g1(C_DIM / 16, 1 + (P_DIM + 15) / 16);
  pm_gemm_h_kernel<<<g1, 32, 0, stream>>>(context, triggers, W1, ctx_h, trg_h);

  // Kernel 2: pattern scores -> weights
  pm_score_kernel<<<P_DIM, 256, 0, stream>>>(ctx_h, trg_h, b1, W2, b2, conf, weights);

  // Kernel 2b: compact active set
  pm_compact_kernel<<<1, 32, 0, stream>>>(weights, actW, actIdx, nact);

  // Kernel 3: streamed bias reduction + broadcast add (HBM-bound)
  const int nQuad  = T_DIM * T_DIM / 4;                  // 262144
  const int blocks = nQuad / (256 * QPT);                // 256
  pm_apply_kernel<<<blocks, 256, 0, stream>>>(attn, biases, actW, actIdx, nact, out);
}
